// ConditionalLayer_53919019434508
// MI455X (gfx1250) — matrix operand order repaired and checked by reading, not compiled, not run on hardware
//
#include <hip/hip_runtime.h>

#define NROWS 16384
#define DDIM  512
#define NCOND 8
#define KBLK  (DDIM / 32)   // 16 K-blocks of 32

typedef float v8f   __attribute__((ext_vector_type(8)));
typedef __bf16 bf16x8 __attribute__((ext_vector_type(8)));
typedef __bf16 v16bf  __attribute__((ext_vector_type(16)));

// Split an fp32 value into bf16 hi (truncated top 16 bits) + bf16 lo (residual).
__device__ __forceinline__ void bsplit(float f, unsigned short& h, unsigned short& l) {
    unsigned u  = __float_as_uint(f);
    unsigned hb = u & 0xffff0000u;
    h = (unsigned short)(hb >> 16);
    float r = f - __uint_as_float(hb);
    l = (unsigned short)(__float_as_uint(r) >> 16);
}

__device__ __forceinline__ v16bf ldfrag(const unsigned short* p0, const unsigned short* p1) {
    bf16x8 a = *(const bf16x8*)p0;
    bf16x8 b = *(const bf16x8*)p1;
    return __builtin_shufflevector(a, b, 0,1,2,3,4,5,6,7,8,9,10,11,12,13,14,15);
}

// ---------------- prep kernels ----------------

__global__ void zero_counts_kernel(int* cnt) {
    if (threadIdx.x < NCOND) cnt[threadIdx.x] = 0;
}

__global__ void bin_kernel(const int* __restrict__ cond, int* cnt, int* rows) {
    int n = blockIdx.x * blockDim.x + threadIdx.x;
    if (n < NROWS) {
        int c = cond[n];
        int i = atomicAdd(&cnt[c], 1);
        rows[c * NROWS + i] = n;
    }
}

// corr[c][d] = sum_c' relu(b[c'][d]) - relu(b[c][d])
__global__ void corr_kernel(const float* __restrict__ b, float* __restrict__ corr) {
    int i = blockIdx.x * blockDim.x + threadIdx.x;   // 0..C*D-1
    int d = i & (DDIM - 1);
    int c = i >> 9;
    float s = 0.f;
#pragma unroll
    for (int c2 = 0; c2 < NCOND; ++c2) {
        float v = b[c2 * DDIM + d];
        s += (v > 0.f) ? v : 0.f;
    }
    float vc = b[c * DDIM + d];
    corr[i] = s - ((vc > 0.f) ? vc : 0.f);
}

// ---------------- grouped GEMM ----------------
// Block: 256 threads (8 waves). Tile: 128 gathered rows x 128 cols, K = 512.
// Double-buffered LDS; bf16 hi/lo split (3-product) for near-fp32 accuracy.
// LDS row stride: 48 bf16 (96B) -> 16B-aligned b128 fragment loads.
#define LSTR 48

__global__ __launch_bounds__(256) void grouped_gemm_kernel(
        const float* __restrict__ x, const float* __restrict__ W,
        const float* __restrict__ bias, const int* __restrict__ cnt,
        const int* __restrict__ rows, const float* __restrict__ corr,
        float* __restrict__ out) {

    __shared__ unsigned short Ahi[2][128 * LSTR];
    __shared__ unsigned short Alo[2][128 * LSTR];
    __shared__ unsigned short Bhi[2][128 * LSTR];   // transposed: [n][k]
    __shared__ unsigned short Blo[2][128 * LSTR];

    const int c       = blockIdx.x >> 7;
    const int tile    = blockIdx.x & 127;
    const int colBase = blockIdx.y * 128;

    const int count = cnt[c];
    const int start = tile * 128;
    if (start >= count) return;
    const int rowsValid = min(128, count - start);

    const int*   rowIdx = rows + c * NROWS + start;
    const float* Wc     = W + (size_t)c * DDIM * DDIM;

    const int t = threadIdx.x;
    // A staging role: row 0..127, half 0..1 (16 floats each)
    const int arow  = t >> 1;
    const int ahalf = t & 1;
    const int grow  = rowIdx[min(arow, rowsValid - 1)];   // clamp padding rows
    const float4* xr = (const float4*)(x + (size_t)grow * DDIM + ahalf * 16);
    // W staging role: k 0..31, 16 consecutive cols
    const int wk  = t >> 3;
    const int wn0 = (t & 7) * 16;

    const int lane = t & 31;
    const int wave = t >> 5;
    const int lrow = lane & 15;
    const int half = lane >> 4;

    v8f acc[8];
#pragma unroll
    for (int i = 0; i < 8; ++i) acc[i] = (v8f)0.f;

    float4 pa[4];   // prefetched x data (16 floats)
    float4 pw[4];   // prefetched W data (16 floats)

    auto loadStage = [&](int kb) {
#pragma unroll
        for (int j = 0; j < 4; ++j) pa[j] = xr[kb * 8 + j];
        const float4* wr = (const float4*)(Wc + (size_t)(kb * 32 + wk) * DDIM + colBase + wn0);
#pragma unroll
        for (int j = 0; j < 4; ++j) pw[j] = wr[j];
    };

    auto storeStage = [&](int buf) {
        unsigned* aph = (unsigned*)&Ahi[buf][arow * LSTR + ahalf * 16];
        unsigned* apl = (unsigned*)&Alo[buf][arow * LSTR + ahalf * 16];
#pragma unroll
        for (int j = 0; j < 4; ++j) {
            unsigned short h0,h1,h2,h3,l0,l1,l2,l3;
            bsplit(pa[j].x,h0,l0); bsplit(pa[j].y,h1,l1);
            bsplit(pa[j].z,h2,l2); bsplit(pa[j].w,h3,l3);
            aph[2*j]   = (unsigned)h0 | ((unsigned)h1 << 16);
            aph[2*j+1] = (unsigned)h2 | ((unsigned)h3 << 16);
            apl[2*j]   = (unsigned)l0 | ((unsigned)l1 << 16);
            apl[2*j+1] = (unsigned)l2 | ((unsigned)l3 << 16);
        }
#pragma unroll
        for (int j = 0; j < 4; ++j) {
            unsigned short h, l;
            bsplit(pw[j].x, h, l); Bhi[buf][(wn0+4*j+0)*LSTR + wk] = h; Blo[buf][(wn0+4*j+0)*LSTR + wk] = l;
            bsplit(pw[j].y, h, l); Bhi[buf][(wn0+4*j+1)*LSTR + wk] = h; Blo[buf][(wn0+4*j+1)*LSTR + wk] = l;
            bsplit(pw[j].z, h, l); Bhi[buf][(wn0+4*j+2)*LSTR + wk] = h; Blo[buf][(wn0+4*j+2)*LSTR + wk] = l;
            bsplit(pw[j].w, h, l); Bhi[buf][(wn0+4*j+3)*LSTR + wk] = h; Blo[buf][(wn0+4*j+3)*LSTR + wk] = l;
        }
    };

    // prologue: fill buffer 0
    loadStage(0);
    storeStage(0);
    __syncthreads();

    for (int kb = 0; kb < KBLK; ++kb) {
        const int cur = kb & 1;

        // issue next block's global loads before the WMMA burst (latency hiding)
        if (kb + 1 < KBLK) loadStage(kb + 1);

        // ---- compute: wave owns rows wave*16..+15, all 128 cols ----
        // A fragment: elems e -> k = half*16 + e (contiguous 16)
        const unsigned short* aBase0 = &Ahi[cur][(wave * 16 + lrow) * LSTR + half * 16];
        const unsigned short* aBase1 = &Alo[cur][(wave * 16 + lrow) * LSTR + half * 16];
        v16bf a_hi = ldfrag(aBase0, aBase0 + 8);
        v16bf a_lo = ldfrag(aBase1, aBase1 + 8);

#pragma unroll
        for (int t0 = 0; t0 < 8; ++t0) {
            const int nb = t0 * 16 + lrow;
            // B fragment: elems e -> k = half*16 + e (contiguous 16)
            const unsigned short* bh = &Bhi[cur][nb * LSTR + half * 16];
            const unsigned short* bl = &Blo[cur][nb * LSTR + half * 16];
            v16bf b_hi = ldfrag(bh, bh + 8);
            v16bf b_lo = ldfrag(bl, bl + 8);
            acc[t0] = __builtin_amdgcn_wmma_f32_16x16x32_bf16(false, a_hi, false, b_hi,
                                                              (short)0, acc[t0], false, false);
            acc[t0] = __builtin_amdgcn_wmma_f32_16x16x32_bf16(false, a_hi, false, b_lo,
                                                              (short)0, acc[t0], false, false);
            acc[t0] = __builtin_amdgcn_wmma_f32_16x16x32_bf16(false, a_lo, false, b_hi,
                                                              (short)0, acc[t0], false, false);
        }

        // convert + store next block into the other buffer, then one barrier
        if (kb + 1 < KBLK) storeStage(1 - cur);
        __syncthreads();
    }

    // ---- epilogue: out[row] = relu(acc + b_c) + corr_c, scattered by row index ----
    int gr[8];
#pragma unroll
    for (int r = 0; r < 8; ++r) {
        int ml = wave * 16 + half * 8 + r;        // C/D layout: M = r + 8*half
        gr[r] = (ml < rowsValid) ? rowIdx[ml] : -1;
    }
#pragma unroll
    for (int t0 = 0; t0 < 8; ++t0) {
        const int col = colBase + t0 * 16 + lrow; // C/D layout: N = lane%16
        const float bv = bias[c * DDIM + col];
        const float cv = corr[c * DDIM + col];
#pragma unroll
        for (int r = 0; r < 8; ++r) {
            if (gr[r] >= 0) {
                float v = acc[t0][r] + bv;
                v = (v > 0.f) ? v : 0.f;
                out[(size_t)gr[r] * DDIM + col] = v + cv;
            }
        }
    }
}

// ---------------- launcher ----------------

extern "C" void kernel_launch(void* const* d_in, const int* in_sizes, int n_in,
                              void* d_out, int out_size, void* d_ws, size_t ws_size,
                              hipStream_t stream) {
    const float* x    = (const float*)d_in[0];
    const int*   cond = (const int*)d_in[1];
    const float* W    = (const float*)d_in[2];
    const float* b    = (const float*)d_in[3];
    float* out = (float*)d_out;

    // workspace layout
    int*   cnt  = (int*)d_ws;                                     // C ints
    float* corr = (float*)((char*)d_ws + 1024);                   // C*D floats
    int*   rows = (int*)((char*)d_ws + 1024 + NCOND * DDIM * 4);  // C*N ints

    zero_counts_kernel<<<1, 32, 0, stream>>>(cnt);
    bin_kernel<<<NROWS / 256, 256, 0, stream>>>(cond, cnt, rows);
    corr_kernel<<<(NCOND * DDIM) / 256, 256, 0, stream>>>(b, corr);

    dim3 grid(NCOND * (NROWS / 128), DDIM / 128);
    grouped_gemm_kernel<<<grid, 256, 0, stream>>>(x, W, b, cnt, rows, corr, out);
}
